// PointNet2SSGSemanticSegmentation_46420006535337
// MI455X (gfx1250) — matrix operand order repaired and checked by reading, not compiled, not run on hardware
//
#include <hip/hip_runtime.h>

#define BATCH 8

typedef __attribute__((ext_vector_type(16))) _Float16 v16h;
typedef __attribute__((ext_vector_type(8)))  _Float16 v8h;
typedef __attribute__((ext_vector_type(8)))  float    v8f;

// ---------------------------------------------------------------------------
// A-fragment builder: 16-bit A 16x32 layout. Lane row = o, K-chunks at
// hi*16 (halves 0..7) and hi*16+8 (halves 8..15).
// ---------------------------------------------------------------------------
__device__ __forceinline__ v16h frag_a(const _Float16* wrow_c0, int hi) {
  v8h lo = *(const v8h*)(wrow_c0 + hi * 16);
  v8h hh = *(const v8h*)(wrow_c0 + hi * 16 + 8);
  return __builtin_shufflevector(lo, hh, 0, 1, 2, 3, 4, 5, 6, 7, 8, 9, 10, 11,
                                 12, 13, 14, 15);
}

// ---------------------------------------------------------------------------
// Branch-free WMMA GEMM, 2x2 register tiling (32 points x 32 channels/wave).
//   Wp : [Op, Cp] f16 zero-padded, X: [M, Cp] f16, Y: [M, Op] f16
//   Y[m,o] = sum_c X[m,c]*Wp[o,c] + bias[o]
// Per k-step: 2 A-fragments (W rows) + 2 B-fragments (X rows, contiguous
// v16h) -> 4 x v_wmma_f32_16x16x32_f16. Op % 32 == 0, Cp % 32 == 0,
// M % 256 == 0. EXEC all-1s throughout.
// ---------------------------------------------------------------------------
__global__ void gemm_wmma_kernel(const _Float16* __restrict__ Wp,
                                 const float* __restrict__ bias,
                                 const _Float16* __restrict__ X,
                                 _Float16* __restrict__ Y,
                                 int Op, int Cp, int M) {
  const int wave = threadIdx.x >> 5;
  const int lane = threadIdx.x & 31;
  const int m0 = (blockIdx.x * 8 + wave) * 32;
  const int o0 = blockIdx.y * 32;
  if (m0 >= M) return;  // never taken; wave-uniform
  const int l16 = lane & 15;
  const int hi  = lane >> 4;

  const _Float16* wrow0 = Wp + (size_t)(o0 + l16) * Cp;
  const _Float16* wrow1 = wrow0 + (size_t)16 * Cp;
  const _Float16* xrow0 = X + (size_t)(m0 + l16) * Cp;
  const _Float16* xrow1 = xrow0 + (size_t)16 * Cp;

  v8f acc00 = {}, acc01 = {}, acc10 = {}, acc11 = {};
  for (int c0 = 0; c0 < Cp; c0 += 32) {
    __builtin_prefetch((const void*)(xrow0 + c0 + 64), 0, 1);
    __builtin_prefetch((const void*)(xrow1 + c0 + 64), 0, 1);
    v16h a0 = frag_a(wrow0 + c0, hi);
    v16h a1 = frag_a(wrow1 + c0, hi);
    v16h b0 = *(const v16h*)(xrow0 + c0 + hi * 16);
    v16h b1 = *(const v16h*)(xrow1 + c0 + hi * 16);
    acc00 = __builtin_amdgcn_wmma_f32_16x16x32_f16(false, a0, false, b0,
                                                   (short)0, acc00, false,
                                                   false);
    acc01 = __builtin_amdgcn_wmma_f32_16x16x32_f16(false, a0, false, b1,
                                                   (short)0, acc01, false,
                                                   false);
    acc10 = __builtin_amdgcn_wmma_f32_16x16x32_f16(false, a1, false, b0,
                                                   (short)0, acc10, false,
                                                   false);
    acc11 = __builtin_amdgcn_wmma_f32_16x16x32_f16(false, a1, false, b1,
                                                   (short)0, acc11, false,
                                                   false);
  }
  float bv0[8], bv1[8];
#pragma unroll
  for (int r = 0; r < 8; ++r) {
    bv0[r] = bias[o0 + hi * 8 + r];
    bv1[r] = bias[o0 + 16 + hi * 8 + r];
  }
  v8h s00, s01, s10, s11;
#pragma unroll
  for (int r = 0; r < 8; ++r) {
    s00[r] = (_Float16)(acc00[r] + bv0[r]);
    s01[r] = (_Float16)(acc01[r] + bv0[r]);
    s10[r] = (_Float16)(acc10[r] + bv1[r]);
    s11[r] = (_Float16)(acc11[r] + bv1[r]);
  }
  _Float16* yr0 = Y + (size_t)(m0 + l16) * Op;
  _Float16* yr1 = Y + (size_t)(m0 + 16 + l16) * Op;
  *(v8h*)(yr0 + o0 + hi * 8) = s00;        // o-tile0, points m0..m0+15
  *(v8h*)(yr1 + o0 + hi * 8) = s01;        // o-tile0, points m0+16..
  *(v8h*)(yr0 + o0 + 16 + hi * 8) = s10;   // o-tile1, points m0..
  *(v8h*)(yr1 + o0 + 16 + hi * 8) = s11;   // o-tile1, points m0+16..
}

// ---------------------------------------------------------------------------
// Pack f32 weights [O, C] -> zero-padded f16 [Op, Cp]; pad bias to [Op].
// ---------------------------------------------------------------------------
__global__ void pack_w_kernel(const float* __restrict__ W,
                              _Float16* __restrict__ Wp, int O, int C, int Cp,
                              int total) {
  int g = blockIdx.x * blockDim.x + threadIdx.x;
  if (g >= total) return;
  int o = g / Cp, c = g % Cp;
  float v = (o < O && c < C) ? W[(size_t)o * C + c] : 0.f;
  Wp[g] = (_Float16)v;
}

__global__ void pack_bias_kernel(const float* __restrict__ b,
                                 float* __restrict__ bp, int O, int Op) {
  int g = blockIdx.x * blockDim.x + threadIdx.x;
  if (g >= Op) return;
  bp[g] = (g < O) ? b[g] : 0.f;
}

// ---------------------------------------------------------------------------
// FPS: one block/batch, serial S steps. Per-step argmax = wave32 __shfl_xor
// butterfly + tiny LDS merge of the 8 wave winners (3 barriers per step).
// Ties resolve to lowest index (jnp.argmax semantics).
// ---------------------------------------------------------------------------
template <int NPT>
__global__ void fps_kernel(const float* __restrict__ coords, int N, int S,
                           int* __restrict__ out_idx) {
  const int b = blockIdx.x;
  const int t = threadIdx.x;
  const float* cx = coords + (size_t)b * 3 * N;
  const float* cy = cx + N;
  const float* cz = cy + N;

  float px[NPT], py[NPT], pz[NPT], dmin[NPT];
#pragma unroll
  for (int i = 0; i < NPT; ++i) {
    int p = t + i * 256;
    bool ok = p < N;
    px[i] = ok ? cx[p] : 0.f;
    py[i] = ok ? cy[p] : 0.f;
    pz[i] = ok ? cz[p] : 0.f;
    dmin[i] = ok ? 1e10f : -1.f;
  }

  __shared__ float s_wval[8];
  __shared__ int s_widx[8];
  __shared__ float s_far[3];
  __shared__ int s_fidx;
  if (t == 0) s_fidx = 0;
  __syncthreads();

  for (int s = 0; s < S; ++s) {
    if (t == 0) {
      int far = s_fidx;
      out_idx[b * S + s] = far;
      s_far[0] = cx[far];
      s_far[1] = cy[far];
      s_far[2] = cz[far];
    }
    __syncthreads();
    float fx = s_far[0], fy = s_far[1], fz = s_far[2];
    float best = -2.f;
    int bestp = 0;
#pragma unroll
    for (int i = 0; i < NPT; ++i) {
      if (dmin[i] >= 0.f) {
        float dx = px[i] - fx, dy = py[i] - fy, dz = pz[i] - fz;
        float d = dx * dx + dy * dy + dz * dz;
        dmin[i] = fminf(dmin[i], d);
      }
      if (dmin[i] > best) { best = dmin[i]; bestp = t + i * 256; }
    }
    // wave32 butterfly argmax
#pragma unroll
    for (int off = 16; off > 0; off >>= 1) {
      float ov = __shfl_xor(best, off, 32);
      int oi = __shfl_xor(bestp, off, 32);
      if (ov > best || (ov == best && oi < bestp)) { best = ov; bestp = oi; }
    }
    if ((t & 31) == 0) {
      s_wval[t >> 5] = best;
      s_widx[t >> 5] = bestp;
    }
    __syncthreads();
    if (t == 0) {
      float bv = s_wval[0];
      int bi = s_widx[0];
#pragma unroll
      for (int w = 1; w < 8; ++w) {
        float ov = s_wval[w];
        int oi = s_widx[w];
        if (ov > bv || (ov == bv && oi < bi)) { bv = ov; bi = oi; }
      }
      s_fidx = bi;
    }
    __syncthreads();
  }
}

// ---------------------------------------------------------------------------
__global__ void gather_coords_kernel(const float* __restrict__ co,
                                     const int* __restrict__ idx,
                                     float* __restrict__ nc, int N, int S) {
  int g = blockIdx.x * blockDim.x + threadIdx.x;
  if (g >= BATCH * S) return;
  int b = g / S, s = g % S;
  int p = idx[b * S + s];
  for (int c = 0; c < 3; ++c)
    nc[(b * 3 + c) * S + s] = co[(size_t)(b * 3 + c) * N + p];
}

// ---------------------------------------------------------------------------
// Ball query: top-32 KNN by insertion, candidate lists in LDS with a
// conflict-free [j*256 + tid] layout (f32 dist + u16 idx = 48 KB / block).
// Radius fallback to the single nearest neighbor.
// ---------------------------------------------------------------------------
__global__ void ball_query_kernel(const float* __restrict__ nc,
                                  const float* __restrict__ co, int N, int S,
                                  float r2, int* __restrict__ out) {
  const int K = 32;
  __shared__ float s_td[K * 256];
  __shared__ unsigned short s_ti[K * 256];
  const int t = threadIdx.x;
  int g = blockIdx.x * 256 + t;
  bool active = g < BATCH * S;
  int b = active ? g / S : 0;
  int s = active ? g % S : 0;

  float qx = nc[(b * 3 + 0) * S + s];
  float qy = nc[(b * 3 + 1) * S + s];
  float qz = nc[(b * 3 + 2) * S + s];
  const float* cx = co + (size_t)b * 3 * N;
  const float* cy = cx + N;
  const float* cz = cy + N;

#pragma unroll
  for (int j = 0; j < K; ++j) {
    s_td[j * 256 + t] = 3.4e38f;
    s_ti[j * 256 + t] = 0;
  }
  for (int n = 0; n < N; ++n) {
    float dx = cx[n] - qx, dy = cy[n] - qy, dz = cz[n] - qz;
    float d = dx * dx + dy * dy + dz * dz;
    if (d < s_td[(K - 1) * 256 + t]) {
      int j = K - 1;
      while (j > 0 && s_td[(j - 1) * 256 + t] > d) {
        s_td[j * 256 + t] = s_td[(j - 1) * 256 + t];
        s_ti[j * 256 + t] = s_ti[(j - 1) * 256 + t];
        --j;
      }
      s_td[j * 256 + t] = d;
      s_ti[j * 256 + t] = (unsigned short)n;
    }
  }
  if (active) {
    int nearest = s_ti[t];
    for (int k = 0; k < K; ++k) {
      float dk = s_td[k * 256 + t];
      int ik = s_ti[k * 256 + t];
      out[(size_t)g * K + k] = (dk > r2) ? nearest : ik;
    }
  }
}

// ---------------------------------------------------------------------------
// Grouped GEMM input X[B*S*32, Cp] f16, elementwise (coalesced):
//   c<3: gathered coords - center;  3..3+Cf: features;  tail: zero pad.
// ---------------------------------------------------------------------------
__global__ void group_kernel(const float* __restrict__ co,
                             const float* __restrict__ nc,
                             const _Float16* __restrict__ feat,
                             const int* __restrict__ idx, int N, int S, int Cf,
                             int Cfp, int Cp, _Float16* __restrict__ X,
                             int total) {
  int g = blockIdx.x * blockDim.x + threadIdx.x;
  if (g >= total) return;
  int m = g / Cp, c = g % Cp;
  int bs = m >> 5;
  int s = bs % S, b = bs / S;
  int p = idx[m];
  float v;
  if (c < 3)
    v = co[(size_t)(b * 3 + c) * N + p] - nc[(b * 3 + c) * S + s];
  else if (c < 3 + Cf)
    v = (float)feat[(size_t)(b * N + p) * Cfp + (c - 3)];
  else
    v = 0.f;
  X[g] = (_Float16)v;
}

// ---------------------------------------------------------------------------
// BN statistics, coalesced: elementwise pass accumulating per-channel
// sum/sumsq in LDS (ds_add_f32), flushed with global atomics; finalize
// computes mean + biased var. Op is a power of two (32..512).
// ---------------------------------------------------------------------------
__global__ void zero_f_kernel(float* __restrict__ p, int n) {
  int g = blockIdx.x * blockDim.x + threadIdx.x;
  if (g < n) p[g] = 0.f;
}

__global__ void bn_stats_partial_kernel(const _Float16* __restrict__ Y,
                                        int total, int Op,
                                        float* __restrict__ acc) {
  __shared__ float lsum[512], lsum2[512];
  for (int c = threadIdx.x; c < Op; c += 256) {
    lsum[c] = 0.f;
    lsum2[c] = 0.f;
  }
  __syncthreads();
  int base = blockIdx.x * 256 * 32 + threadIdx.x;
  for (int i = 0; i < 32; ++i) {
    int e = base + i * 256;
    if (e < total) {
      float v = (float)Y[e];
      int c = e & (Op - 1);
      atomicAdd(&lsum[c], v);
      atomicAdd(&lsum2[c], v * v);
    }
  }
  __syncthreads();
  for (int c = threadIdx.x; c < Op; c += 256) {
    atomicAdd(&acc[c], lsum[c]);
    atomicAdd(&acc[Op + c], lsum2[c]);
  }
}

__global__ void bn_finalize_kernel(const float* __restrict__ acc,
                                   float* __restrict__ stats, int Op,
                                   float invM) {
  int o = blockIdx.x * blockDim.x + threadIdx.x;
  if (o >= Op) return;
  float mean = acc[o] * invM;
  float var = acc[Op + o] * invM - mean * mean;
  stats[2 * o] = mean;
  stats[2 * o + 1] = var;
}

__global__ void bn_relu_kernel(_Float16* __restrict__ Y,
                               const float* __restrict__ stats, int Op,
                               int total) {
  int g = blockIdx.x * blockDim.x + threadIdx.x;
  if (g >= total) return;
  int o = g & (Op - 1);
  float v = ((float)Y[g] - stats[2 * o]) * rsqrtf(stats[2 * o + 1] + 1e-5f);
  Y[g] = (_Float16)fmaxf(v, 0.f);
}

// ---------------------------------------------------------------------------
// Max over the 32 neighbors: Y [BS*32, Op] -> F [BS, Op]
// ---------------------------------------------------------------------------
__global__ void maxpool_kernel(const _Float16* __restrict__ Y,
                               _Float16* __restrict__ F, int Op, int BS) {
  int g = blockIdx.x * blockDim.x + threadIdx.x;
  if (g >= BS * Op) return;
  int bs = g / Op, o = g & (Op - 1);
  const _Float16* p = Y + (size_t)bs * 32 * Op + o;
  float mx = (float)p[0];
#pragma unroll
  for (int k = 1; k < 32; ++k) mx = fmaxf(mx, (float)p[(size_t)k * Op]);
  F[g] = (_Float16)mx;
}

// ---------------------------------------------------------------------------
// FP concat (elementwise): skip-features into row head, zero the pad tail;
// the middle [C1, C1+C2) is filled by fp_gather.
// ---------------------------------------------------------------------------
__global__ void fp_concat_kernel(const _Float16* __restrict__ f1,
                                 _Float16* __restrict__ X, int C1, int C1p,
                                 int C2, int Cp, int total) {
  int g = blockIdx.x * blockDim.x + threadIdx.x;
  if (g >= total) return;
  int m = g / Cp, c = g % Cp;
  if (c < C1)
    X[g] = f1[(size_t)m * C1p + c];
  else if (c >= C1 + C2)
    X[g] = (_Float16)0.f;
}

// ---------------------------------------------------------------------------
// FP 3-NN: per fine point, scan coarse points, store indices + normalized
// inverse-distance weights.
// ---------------------------------------------------------------------------
__global__ void fp_knn_kernel(const float* __restrict__ c1,
                              const float* __restrict__ c2, int N1, int N2,
                              int* __restrict__ nidx,
                              float* __restrict__ nw) {
  int g = blockIdx.x * blockDim.x + threadIdx.x;
  if (g >= BATCH * N1) return;
  int b = g / N1, n = g % N1;
  float qx = c1[(size_t)(b * 3 + 0) * N1 + n];
  float qy = c1[(size_t)(b * 3 + 1) * N1 + n];
  float qz = c1[(size_t)(b * 3 + 2) * N1 + n];
  const float* cx = c2 + (size_t)b * 3 * N2;
  const float* cy = cx + N2;
  const float* cz = cy + N2;

  float d0 = 3.4e38f, d1 = 3.4e38f, d2 = 3.4e38f;
  int i0 = 0, i1 = 0, i2 = 0;
  for (int j = 0; j < N2; ++j) {
    float dx = cx[j] - qx, dy = cy[j] - qy, dz = cz[j] - qz;
    float d = dx * dx + dy * dy + dz * dz;
    if (d < d0) {
      d2 = d1; i2 = i1;
      d1 = d0; i1 = i0;
      d0 = d;  i0 = j;
    } else if (d < d1) {
      d2 = d1; i2 = i1;
      d1 = d;  i1 = j;
    } else if (d < d2) {
      d2 = d;  i2 = j;
    }
  }
  float w0 = 1.f / fmaxf(d0, 1e-10f);
  float w1 = 1.f / fmaxf(d1, 1e-10f);
  float w2 = 1.f / fmaxf(d2, 1e-10f);
  float ws = w0 + w1 + w2;
  nidx[3 * g + 0] = i0;
  nidx[3 * g + 1] = i1;
  nidx[3 * g + 2] = i2;
  nw[3 * g + 0] = w0 / ws;
  nw[3 * g + 1] = w1 / ws;
  nw[3 * g + 2] = w2 / ws;
}

// ---------------------------------------------------------------------------
// FP gather (elementwise): interp channels into X rows at offset C1.
// ---------------------------------------------------------------------------
__global__ void fp_gather_kernel(const _Float16* __restrict__ f2,
                                 const int* __restrict__ nidx,
                                 const float* __restrict__ nw, int N1, int N2,
                                 int C2, int C2p, _Float16* __restrict__ X,
                                 int C1, int Cp, int total) {
  int g = blockIdx.x * blockDim.x + threadIdx.x;
  if (g >= total) return;
  int m = g / C2, c = g % C2;
  int b = m / N1;
  const int* id = nidx + 3 * m;
  const float* w = nw + 3 * m;
  size_t base = (size_t)b * N2;
  float v = w[0] * (float)f2[(base + id[0]) * C2p + c] +
            w[1] * (float)f2[(base + id[1]) * C2p + c] +
            w[2] * (float)f2[(base + id[2]) * C2p + c];
  X[(size_t)m * Cp + C1 + c] = (_Float16)v;
}

// ---------------------------------------------------------------------------
// Glue
// ---------------------------------------------------------------------------
__global__ void repack_feat_kernel(const float* __restrict__ in,
                                   _Float16* __restrict__ out, int Npts,
                                   int total) {
  // [B, 3, N] f32 -> [B*N, 32] f16 zero padded, elementwise
  int g = blockIdx.x * blockDim.x + threadIdx.x;
  if (g >= total) return;
  int m = g >> 5, c = g & 31;
  int b = m / Npts, n = m % Npts;
  out[g] = (_Float16)((c < 3) ? in[((size_t)b * 3 + c) * Npts + n] : 0.f);
}

__global__ void copy_h_kernel(const _Float16* __restrict__ src,
                              _Float16* __restrict__ dst, int n) {
  int g = blockIdx.x * blockDim.x + threadIdx.x;
  if (g < n) dst[g] = src[g];
}

__global__ void unpack_out_kernel(const _Float16* __restrict__ Y,
                                  float* __restrict__ out, int O, int Op,
                                  int Npts) {
  // Y [B*N, Op] -> out [B, O, N] f32
  int g = blockIdx.x * blockDim.x + threadIdx.x;
  if (g >= BATCH * O * Npts) return;
  int n = g % Npts;
  int rest = g / Npts;
  int o = rest % O;
  int b = rest / O;
  out[g] = (float)Y[(size_t)(b * Npts + n) * Op + o];
}

// ---------------------------------------------------------------------------
// Host orchestration
// ---------------------------------------------------------------------------
static inline int cdiv(int a, int b) { return (a + b - 1) / b; }
static inline int pad32(int x) { return (x + 31) & ~31; }

extern "C" void kernel_launch(void* const* d_in, const int* in_sizes, int n_in,
                              void* d_out, int out_size, void* d_ws,
                              size_t ws_size, hipStream_t stream) {
  (void)in_sizes; (void)out_size; (void)ws_size;
  const int N0 = 8192;

  const float* coords = (const float*)d_in[0];
  const float* feats = (const float*)d_in[1];
  const float* p[48] = {nullptr};
  for (int i = 2; i < 48 && i < n_in; ++i) p[i] = (const float*)d_in[i];

  // ---- workspace carve ----
  char* wsp = (char*)d_ws;
  auto alloc_f = [&](size_t n) {
    float* r = (float*)wsp;
    wsp += ((n * sizeof(float) + 255) & ~(size_t)255);
    return r;
  };
  auto alloc_h = [&](size_t n) {
    _Float16* r = (_Float16*)wsp;
    wsp += ((n * sizeof(_Float16) + 255) & ~(size_t)255);
    return r;
  };
  auto alloc_i = [&](size_t n) {
    int* r = (int*)wsp;
    wsp += ((n * sizeof(int) + 255) & ~(size_t)255);
    return r;
  };

  _Float16* feat0 = alloc_h((size_t)BATCH * N0 * 32);  // [B*N, 32]
  _Float16* ping = alloc_h((size_t)16777216);          // 262144 x 64 max
  _Float16* pong = alloc_h((size_t)16777216);
  float* l1c = alloc_f((size_t)BATCH * 3 * 1024);
  float* l2c = alloc_f((size_t)BATCH * 3 * 256);
  float* l3c = alloc_f((size_t)BATCH * 3 * 64);
  float* l4c = alloc_f((size_t)BATCH * 3 * 16);
  _Float16* l1f = alloc_h((size_t)BATCH * 1024 * 64);
  _Float16* l2f = alloc_h((size_t)BATCH * 256 * 128);
  _Float16* l3f = alloc_h((size_t)BATCH * 64 * 256);
  _Float16* l4f = alloc_h((size_t)BATCH * 16 * 512);
  _Float16* l3fn = alloc_h((size_t)BATCH * 64 * 256);
  _Float16* l2fn = alloc_h((size_t)BATCH * 256 * 256);
  _Float16* l1fn = alloc_h((size_t)BATCH * 1024 * 128);
  _Float16* Wp = alloc_h((size_t)512 * 768);
  float* biasp = alloc_f(512);
  float* stats = alloc_f(1024);
  float* bnacc = alloc_f(1024);
  int* fps_idx = alloc_i((size_t)BATCH * 1024);
  int* bq_idx = alloc_i((size_t)BATCH * 1024 * 32);
  int* nidx = alloc_i((size_t)BATCH * N0 * 3);
  float* nw = alloc_f((size_t)BATCH * N0 * 3);

  // ---- helpers ----
  auto mlp_layer = [&](const float* W, const float* bias, int O, int Op,
                       int Creal, int Cp, const _Float16* X, _Float16* Y,
                       int M, bool bn) {
    int wtot = Op * Cp;
    pack_w_kernel<<<dim3(cdiv(wtot, 256)), dim3(256), 0, stream>>>(
        W, Wp, O, Creal, Cp, wtot);
    pack_bias_kernel<<<dim3(cdiv(Op, 256)), dim3(256), 0, stream>>>(bias,
                                                                    biasp, O,
                                                                    Op);
    gemm_wmma_kernel<<<dim3(M / 256, Op / 32), dim3(256), 0, stream>>>(
        Wp, biasp, X, Y, Op, Cp, M);
    if (bn) {
      int tot = M * Op;
      zero_f_kernel<<<dim3(cdiv(2 * Op, 256)), dim3(256), 0, stream>>>(bnacc,
                                                                       2 * Op);
      bn_stats_partial_kernel<<<dim3(cdiv(tot, 8192)), dim3(256), 0, stream>>>(
          Y, tot, Op, bnacc);
      bn_finalize_kernel<<<dim3(cdiv(Op, 256)), dim3(256), 0, stream>>>(
          bnacc, stats, Op, 1.f / (float)M);
      bn_relu_kernel<<<dim3(cdiv(tot, 256)), dim3(256), 0, stream>>>(Y, stats,
                                                                     Op, tot);
    }
  };

  auto run_fps = [&](const float* co, int N, int S) {
    if (N >= 8192)
      fps_kernel<32><<<dim3(BATCH), dim3(256), 0, stream>>>(co, N, S, fps_idx);
    else if (N >= 1024)
      fps_kernel<4><<<dim3(BATCH), dim3(256), 0, stream>>>(co, N, S, fps_idx);
    else
      fps_kernel<1><<<dim3(BATCH), dim3(256), 0, stream>>>(co, N, S, fps_idx);
  };

  auto sa_level = [&](const float* co, const _Float16* fin, int Cf, int Cfp,
                      int N, int S, float radius, const float* W0,
                      const float* b0, int O0, const float* W1,
                      const float* b1, int O1, const float* W2,
                      const float* b2, int O2, float* lc, _Float16* lf) {
    run_fps(co, N, S);
    int bs = BATCH * S;
    gather_coords_kernel<<<dim3(cdiv(bs, 256)), dim3(256), 0, stream>>>(
        co, fps_idx, lc, N, S);
    ball_query_kernel<<<dim3(cdiv(bs, 256)), dim3(256), 0, stream>>>(
        lc, co, N, S, radius * radius, bq_idx);
    int M = bs * 32;
    int Cp = pad32(3 + Cf);
    int xtot = M * Cp;
    group_kernel<<<dim3(cdiv(xtot, 256)), dim3(256), 0, stream>>>(
        co, lc, fin, bq_idx, N, S, Cf, Cfp, Cp, ping, xtot);
    mlp_layer(W0, b0, O0, O0, 3 + Cf, Cp, ping, pong, M, true);
    mlp_layer(W1, b1, O1, O1, O0, O0, pong, ping, M, true);
    mlp_layer(W2, b2, O2, O2, O1, O1, ping, pong, M, true);
    maxpool_kernel<<<dim3(cdiv(bs * O2, 256)), dim3(256), 0, stream>>>(
        pong, lf, O2, bs);
  };

  auto fp_level = [&](const float* c1, const float* c2, const _Float16* f1,
                      int C1, int C1p, const _Float16* f2, int C2, int C2p,
                      int N1, int N2, const float* W0, const float* b0, int O0,
                      const float* W1, const float* b1, int O1,
                      _Float16* fout) {
    int M = BATCH * N1;
    int Cp = pad32(C1 + C2);
    int xtot = M * Cp;
    fp_concat_kernel<<<dim3(cdiv(xtot, 256)), dim3(256), 0, stream>>>(
        f1, ping, C1, C1p, C2, Cp, xtot);
    fp_knn_kernel<<<dim3(cdiv(M, 256)), dim3(256), 0, stream>>>(c1, c2, N1, N2,
                                                                nidx, nw);
    int gtot = M * C2;
    fp_gather_kernel<<<dim3(cdiv(gtot, 256)), dim3(256), 0, stream>>>(
        f2, nidx, nw, N1, N2, C2, C2p, ping, C1, Cp, gtot);
    mlp_layer(W0, b0, O0, O0, C1 + C2, Cp, ping, pong, M, true);
    mlp_layer(W1, b1, O1, O1, O0, O0, pong, ping, M, true);
    copy_h_kernel<<<dim3(cdiv(M * O1, 256)), dim3(256), 0, stream>>>(
        ping, fout, M * O1);
  };

  // ---- pipeline ----
  {
    int tot = BATCH * N0 * 32;
    repack_feat_kernel<<<dim3(cdiv(tot, 256)), dim3(256), 0, stream>>>(
        feats, feat0, N0, tot);
  }

  // SA levels
  sa_level(coords, feat0, 3, 32, 8192, 1024, 0.1f, p[2], p[3], 32, p[4], p[5],
           32, p[6], p[7], 64, l1c, l1f);
  sa_level(l1c, l1f, 64, 64, 1024, 256, 0.2f, p[8], p[9], 64, p[10], p[11], 64,
           p[12], p[13], 128, l2c, l2f);
  sa_level(l2c, l2f, 128, 128, 256, 64, 0.4f, p[14], p[15], 128, p[16], p[17],
           128, p[18], p[19], 256, l3c, l3f);
  sa_level(l3c, l3f, 256, 256, 64, 16, 0.8f, p[20], p[21], 256, p[22], p[23],
           256, p[24], p[25], 512, l4c, l4f);

  // FP levels
  fp_level(l3c, l4c, l3f, 256, 256, l4f, 512, 512, 64, 16, p[26], p[27], 256,
           p[28], p[29], 256, l3fn);
  fp_level(l2c, l3c, l2f, 128, 128, l3fn, 256, 256, 256, 64, p[30], p[31], 256,
           p[32], p[33], 256, l2fn);
  fp_level(l1c, l2c, l1f, 64, 64, l2fn, 256, 256, 1024, 256, p[34], p[35], 256,
           p[36], p[37], 128, l1fn);

  // FP1 (3 layers) + head + out
  {
    int M = BATCH * N0;
    int Cp = pad32(3 + 128);  // 160
    int xtot = M * Cp;
    fp_concat_kernel<<<dim3(cdiv(xtot, 256)), dim3(256), 0, stream>>>(
        feat0, ping, 3, 32, 128, Cp, xtot);
    fp_knn_kernel<<<dim3(cdiv(M, 256)), dim3(256), 0, stream>>>(
        coords, l1c, N0, 1024, nidx, nw);
    int gtot = M * 128;
    fp_gather_kernel<<<dim3(cdiv(gtot, 256)), dim3(256), 0, stream>>>(
        l1fn, nidx, nw, N0, 1024, 128, 128, ping, 3, Cp, gtot);
    mlp_layer(p[38], p[39], 128, 128, 131, Cp, ping, pong, M, true);
    mlp_layer(p[40], p[41], 128, 128, 128, 128, pong, ping, M, true);
    mlp_layer(p[42], p[43], 128, 128, 128, 128, ping, pong, M, true);
    // head
    mlp_layer(p[44], p[45], 128, 128, 128, 128, pong, ping, M, true);
    // out conv: O=13 padded to 32, no BN/ReLU
    mlp_layer(p[46], p[47], 13, 32, 128, 128, ping, pong, M, false);
    unpack_out_kernel<<<dim3(cdiv(13 * M, 256)), dim3(256), 0, stream>>>(
        pong, (float*)d_out, 13, 32, N0);
  }
}
